// EnhancedGNN_83571473645715
// MI455X (gfx1250) — matrix operand order repaired and checked by reading, not compiled, not run on hardware
//
#include <hip/hip_runtime.h>
#include <hip/hip_bf16.h>

// EnhancedGNN for MI455X (gfx1250). Dense GEMMs via v_wmma_f32_16x16x32_f16
// with pre-packed B fragments (all-vector loads), edge stages fused with f32
// global atomics, BN via two-pass column stats.

typedef __attribute__((ext_vector_type(16))) _Float16 v16h;
typedef __attribute__((ext_vector_type(8)))  _Float16 v8h;
typedef __attribute__((ext_vector_type(8)))  float    v8f;

#define Hdim   128
#define HEADS  4
#define TWOH   256
#define HH     512   // HEADS*H
#define EDIM   6
#define INDIM  10

// ---------- order-preserving float <-> uint for atomicMax segment-max ----------
__device__ __forceinline__ unsigned fenc(float f) {
  unsigned u = __float_as_uint(f);
  return (u & 0x80000000u) ? ~u : (u | 0x80000000u);
}
__device__ __forceinline__ float fdec(unsigned u) {
  return (u & 0x80000000u) ? __uint_as_float(u ^ 0x80000000u) : __uint_as_float(~u);
}
#define ENC_NEG_INF 0x007FFFFFu   // fenc(-inf)

// ---------------------------- utility kernels ----------------------------
__global__ void fill_f32_kernel(float* p, float v, size_t n) {
  size_t i = blockIdx.x * (size_t)blockDim.x + threadIdx.x;
  size_t s = (size_t)gridDim.x * blockDim.x;
  for (; i < n; i += s) p[i] = v;
}
__global__ void fill_u32_kernel(unsigned* p, unsigned v, size_t n) {
  size_t i = blockIdx.x * (size_t)blockDim.x + threadIdx.x;
  size_t s = (size_t)gridDim.x * blockDim.x;
  for (; i < n; i += s) p[i] = v;
}
__global__ void add_f16_kernel(const float* a, const float* b, _Float16* y, size_t n) {
  size_t i = blockIdx.x * (size_t)blockDim.x + threadIdx.x;
  if (i < n) y[i] = (_Float16)(a[i] + b[i]);
}

// Pack f32 weight B[K,Nc] (row major) into WMMA B-fragment order, f16:
// dest[((kt*tilesN + nt)*32 + lane)*16 + j] = B[kt*32 + (lane>>4)*16 + j][nt*16 + (lane&15)]
// so each lane's 16-half fragment is contiguous (two aligned b128 loads in the GEMM).
__global__ void pack_b_kernel(const float* B, _Float16* Bp, int K, int Nc) {
  size_t idx = blockIdx.x * (size_t)blockDim.x + threadIdx.x;
  size_t n = (size_t)K * Nc;
  size_t stride = (size_t)gridDim.x * blockDim.x;
  int tilesN = Nc >> 4;
  for (; idx < n; idx += stride) {
    int k = (int)(idx / Nc), c = (int)(idx % Nc);
    int kt = k >> 5, kk = k & 31;
    int half = kk >> 4, j = kk & 15;
    int nt = c >> 4, ln = c & 15;
    int lane = half * 16 + ln;
    size_t dest = (((size_t)(kt * tilesN + nt) * 32) + lane) * 16 + j;
    Bp[dest] = (_Float16)B[idx];
  }
}

// ---------------------------- BatchNorm helpers ----------------------------
// stats[0..C) = per-channel sum, stats[C..2C) = per-channel sum of squares
__global__ void col_stats_kernel(const float* X, float* stats, int Nrows, int C, int chunk) {
  int nrb = (Nrows + chunk - 1) / chunk;
  int tid = blockIdx.x * blockDim.x + threadIdx.x;
  if (tid >= C * nrb) return;
  int c = tid % C, rb = tid / C;
  int r0 = rb * chunk;
  int r1 = r0 + chunk; if (r1 > Nrows) r1 = Nrows;
  float s = 0.f, q = 0.f;
  for (int r = r0; r < r1; ++r) {
    float v = X[(size_t)r * C + c];
    s += v; q += v * v;
  }
  atomicAdd(&stats[c], s);
  atomicAdd(&stats[C + c], q);
}
__global__ void bn_finalize_kernel(const float* stats, float* mv, int Nrows, int C) {
  int c = blockIdx.x * blockDim.x + threadIdx.x;
  if (c >= C) return;
  float mean = stats[c] / (float)Nrows;
  float var  = stats[C + c] / (float)Nrows - mean * mean;
  mv[c] = mean;
  mv[C + c] = rsqrtf(var + 1e-5f);
}
// Y (f32, optional) and Yh (f16, optional) outputs; in-place (Y==X) is safe.
__global__ void bn_apply_kernel(const float* X, float* Y, _Float16* Yh, const float* mv,
                                const float* gamma, const float* beta,
                                size_t n, int C, int relu_flag) {
  size_t i = blockIdx.x * (size_t)blockDim.x + threadIdx.x;
  if (i >= n) return;
  int c = (int)(i % C);
  float v = gamma[c] * (X[i] - mv[c]) * mv[C + c] + beta[c];
  if (relu_flag) v = v > 0.f ? v : 0.f;
  if (Y)  Y[i]  = v;
  if (Yh) Yh[i] = (_Float16)v;
}

// ---------------------------- input projection (K=10) ----------------------------
__global__ void ip_proj_kernel(const float* x, const float* w, const float* b, float* z, int N) {
  size_t idx = blockIdx.x * (size_t)blockDim.x + threadIdx.x;
  if (idx >= (size_t)N * Hdim) return;
  int r = (int)(idx / Hdim), c = (int)(idx % Hdim);
  float acc = b[c];
#pragma unroll
  for (int k = 0; k < INDIM; ++k) acc += x[(size_t)r * INDIM + k] * w[k * Hdim + c];
  z[idx] = acc;
}

// ---------------------------- GINE fused edge message + scatter ----------------------------
__global__ void gine_scatter_kernel(const float* h, const float* ea, const float* ew,
                                    const float* eb, const int* src, const int* dst,
                                    float* aggr, int E) {
  int e = blockIdx.x;
  if (e >= E) return;
  int c = threadIdx.x;           // 128 threads = H channels
  int s = src[e], d = dst[e];
  float v = eb[c];
#pragma unroll
  for (int k = 0; k < EDIM; ++k) v += ea[(size_t)e * EDIM + k] * ew[k * Hdim + c];
  v += h[(size_t)s * Hdim + c];
  v = v > 0.f ? v : 0.f;
  atomicAdd(&aggr[(size_t)d * Hdim + c], v);
}

// ---------------------------- WMMA GEMM ----------------------------
// C[M,Nc] = A[M,K](f16,row-major) * B(f16, pre-packed fragments) + bias.
// One wave computes a 16x32 output slab (two 16x16 N-tiles) so each A fragment
// feeds two v_wmma_f32_16x16x32_f16; K unrolled by 64 -> 4 WMMA per iteration.
// Fragment layouts (wave32); A/B k order is this code's contiguous choice, C/D per CDNA5 ISA 7.12.2:
//   A 16x32: lane M = lane&15, half = lane>>4; elem j -> K = half*16 + j
//            (one contiguous 32B run, two 16B loads)
//   B 32x16: lane N = lane&15, half = lane>>4; elem j -> row K = half*16 + j
//            (pre-packed contiguous per lane by pack_b_kernel)
//   C/D 16x16: VGPR i -> row M = half*8 + i, col N = lane&15
__global__ void wmma_gemm_kernel(const _Float16* __restrict__ A, const _Float16* __restrict__ Bp,
                                 const float* __restrict__ bias, float* __restrict__ C,
                                 int M, int K, int Nc) {
  int gthread = blockIdx.x * blockDim.x + threadIdx.x;
  int wave = gthread >> 5;
  int lane = threadIdx.x & 31;
  int tilesN  = Nc >> 4;
  int tilesN2 = Nc >> 5;                    // N-tile pairs
  int tilesM  = (M + 15) >> 4;
  if (wave >= tilesM * tilesN2) return;     // uniform per wave: EXEC stays all-ones
  int tileM = wave / tilesN2;
  int tileN = (wave % tilesN2) * 2;

  int half = lane >> 4;
  int mrow = tileM * 16 + (lane & 15);
  int arow = mrow < M ? mrow : M - 1;       // clamp (keeps EXEC full through WMMA)
  const _Float16* Arow = A + (size_t)arow * K;

  v8f acc0 = {}, acc1 = {};
  for (int kb = 0; kb < K; kb += 64) {
    if (kb + 64 < K) __builtin_prefetch(Arow + kb + 64, 0, 1);   // global_prefetch_b8
#pragma unroll
    for (int u = 0; u < 2; ++u) {
      int k0 = kb + u * 32;
      int kt = k0 >> 5;
      const _Float16* Ap = Arow + k0 + half * 16;
      v8h alo = *(const v8h*)(Ap);
      v8h ahi = *(const v8h*)(Ap + 8);
      v16h a = __builtin_shufflevector(alo, ahi, 0, 1, 2, 3, 4, 5, 6, 7,
                                       8, 9, 10, 11, 12, 13, 14, 15);

      const _Float16* B0 = Bp + (((size_t)(kt * tilesN + tileN) * 32) + lane) * 16;
      v8h b0lo = *(const v8h*)(B0);
      v8h b0hi = *(const v8h*)(B0 + 8);
      v8h b1lo = *(const v8h*)(B0 + 512);       // next N-tile: +32 lanes * 16 halves
      v8h b1hi = *(const v8h*)(B0 + 520);
      v16h b0 = __builtin_shufflevector(b0lo, b0hi, 0, 1, 2, 3, 4, 5, 6, 7,
                                        8, 9, 10, 11, 12, 13, 14, 15);
      v16h b1 = __builtin_shufflevector(b1lo, b1hi, 0, 1, 2, 3, 4, 5, 6, 7,
                                        8, 9, 10, 11, 12, 13, 14, 15);

      acc0 = __builtin_amdgcn_wmma_f32_16x16x32_f16(false, a, false, b0,
                                                    (short)0, acc0, false, false);
      acc1 = __builtin_amdgcn_wmma_f32_16x16x32_f16(false, a, false, b1,
                                                    (short)0, acc1, false, false);
    }
  }

  int col0 = tileN * 16 + (lane & 15);
  float bv0 = bias ? bias[col0] : 0.f;
  float bv1 = bias ? bias[col0 + 16] : 0.f;
  int rbase = tileM * 16 + half * 8;
  if (tileM * 16 + 16 <= M) {               // fast path: no row guards
#pragma unroll
    for (int i = 0; i < 8; ++i) {
      float* Cr = C + (size_t)(rbase + i) * Nc + col0;
      Cr[0]  = acc0[i] + bv0;
      Cr[16] = acc1[i] + bv1;
    }
  } else {
#pragma unroll
    for (int i = 0; i < 8; ++i) {
      int r = rbase + i;
      if (r < M) {
        float* Cr = C + (size_t)r * Nc + col0;
        Cr[0]  = acc0[i] + bv0;
        Cr[16] = acc1[i] + bv1;
      }
    }
  }
}

// ---------------------------- skip connections ----------------------------
__global__ void skip0_kernel(const float* h0, float* z2, const float* skip, size_t n) {
  size_t i = blockIdx.x * (size_t)blockDim.x + threadIdx.x;
  if (i >= n) return;
  float s = 1.f / (1.f + __expf(-skip[0]));
  float v = z2[i]; v = v > 0.f ? v : 0.f;
  z2[i] = s * h0[i] + (1.f - s) * v;
}

// ---------------------------- edge_attr mean -> self-loop edge embedding ----------------------------
__global__ void ea_sum_kernel(const float* ea, float* sum6, int E) {
  size_t tid = blockIdx.x * (size_t)blockDim.x + threadIdx.x;
  size_t stride = (size_t)gridDim.x * blockDim.x;
  float s[EDIM] = {0.f, 0.f, 0.f, 0.f, 0.f, 0.f};
  for (size_t e = tid; e < (size_t)E; e += stride)
#pragma unroll
    for (int k = 0; k < EDIM; ++k) s[k] += ea[e * EDIM + k];
#pragma unroll
  for (int k = 0; k < EDIM; ++k) atomicAdd(&sum6[k], s[k]);
}
__global__ void ee_mean_kernel(const float* sum6, const float* we, float* eemean, int E) {
  int c = blockIdx.x * blockDim.x + threadIdx.x;
  if (c >= HH) return;
  float v = 0.f;
#pragma unroll
  for (int k = 0; k < EDIM; ++k) v += (sum6[k] / (float)E) * we[k * HH + c];
  eemean[c] = v;
}

// ---------------------------- GATv2 attention ----------------------------
// Pass 1: per (edge, head) logit + segment max. blockDim = 128 (4 heads x 32 lanes).
__global__ void gat_logits_kernel(const float* xl, const float* xr, const float* ea,
                                  const float* we, const float* att, const float* eemean,
                                  const int* src, const int* dst,
                                  float* logits, unsigned* lmaxu,
                                  int count, int loop_mode, int base) {
  int e = blockIdx.x;
  if (e >= count) return;
  int t = threadIdx.x, h = t >> 5, l = t & 31;
  int s = loop_mode ? e : src[e];
  int d = loop_mode ? e : dst[e];
  float eav[EDIM];
  if (!loop_mode) {
#pragma unroll
    for (int k = 0; k < EDIM; ++k) eav[k] = ea[(size_t)e * EDIM + k];
  }
  float p = 0.f;
  for (int dd = l; dd < Hdim; dd += 32) {
    int c = h * Hdim + dd;
    float ee;
    if (loop_mode) ee = eemean[c];
    else {
      ee = 0.f;
#pragma unroll
      for (int k = 0; k < EDIM; ++k) ee += eav[k] * we[k * HH + c];
    }
    float m = xl[(size_t)s * HH + c] + xr[(size_t)d * HH + c] + ee;
    m = m > 0.f ? m : 0.2f * m;          // leaky_relu(0.2)
    p += m * att[c];
  }
  for (int off = 16; off > 0; off >>= 1) p += __shfl_down(p, off, 32);
  if (l == 0) {
    logits[((size_t)(base + e)) * HEADS + h] = p;
    atomicMax(&lmaxu[(size_t)d * HEADS + h], fenc(p));
  }
}

// Pass 2: a = exp(logit - max), accumulate denominator. (in-place overwrite of logits)
__global__ void gat_softmax_kernel(float* logits, const unsigned* lmaxu, float* denom,
                                   const int* dst, int E, int N) {
  size_t idx = blockIdx.x * (size_t)blockDim.x + threadIdx.x;
  size_t total = (size_t)(E + N) * HEADS;
  if (idx >= total) return;
  int item = (int)(idx >> 2), h = (int)(idx & 3);
  int d = item < E ? dst[item] : (item - E);
  float a = __expf(logits[idx] - fdec(lmaxu[(size_t)d * HEADS + h]));
  logits[idx] = a;
  atomicAdd(&denom[(size_t)d * HEADS + h], a);
}

// Pass 3: weighted scatter of xl[src] by alpha. blockDim = 128 (d-channel), loop heads.
__global__ void gat_aggregate_kernel(const float* xl, const float* a, const float* denom,
                                     const int* src, const int* dst, float* accum,
                                     int count, int loop_mode, int base) {
  int e = blockIdx.x;
  if (e >= count) return;
  int s = loop_mode ? e : src[e];
  int d = loop_mode ? e : dst[e];
  int t = threadIdx.x;
#pragma unroll
  for (int h = 0; h < HEADS; ++h) {
    float alpha = a[((size_t)(base + e)) * HEADS + h] /
                  (denom[(size_t)d * HEADS + h] + 1e-16f);
    atomicAdd(&accum[(size_t)d * HH + h * Hdim + t],
              xl[(size_t)s * HH + h * Hdim + t] * alpha);
  }
}

// head mean + bias + relu + skip1
__global__ void gat_combine_kernel(const float* accum, const float* h1, const float* gbias,
                                   const float* skip, float* h2, int N) {
  size_t idx = blockIdx.x * (size_t)blockDim.x + threadIdx.x;
  if (idx >= (size_t)N * Hdim) return;
  int i = (int)(idx / Hdim), c = (int)(idx % Hdim);
  float g = 0.25f * (accum[(size_t)i * HH + 0 * Hdim + c] +
                     accum[(size_t)i * HH + 1 * Hdim + c] +
                     accum[(size_t)i * HH + 2 * Hdim + c] +
                     accum[(size_t)i * HH + 3 * Hdim + c]) + gbias[c];
  g = g > 0.f ? g : 0.f;
  float s = 1.f / (1.f + __expf(-skip[0]));
  h2[idx] = s * h1[idx] + (1.f - s) * g;
}

__global__ void final_avg_kernel(const float* h0, const float* h1, const float* h2,
                                 float* out, size_t n) {
  size_t i = blockIdx.x * (size_t)blockDim.x + threadIdx.x;
  if (i < n) out[i] = (h0[i] + h1[i] + h2[i]) * (1.f / 3.f);
}

// ============================== host launch ==============================
static inline int nb(size_t n, int t) { return (int)((n + t - 1) / t); }

extern "C" void kernel_launch(void* const* d_in, const int* in_sizes, int n_in,
                              void* d_out, int out_size, void* d_ws, size_t ws_size,
                              hipStream_t stream) {
  const float* x          = (const float*)d_in[0];
  const float* edge_attr  = (const float*)d_in[1];
  const float* ip_w       = (const float*)d_in[2];
  const float* ip_b       = (const float*)d_in[3];
  const float* ip_gamma   = (const float*)d_in[4];
  const float* ip_beta    = (const float*)d_in[5];
  const float* gine_ew    = (const float*)d_in[6];
  const float* gine_eb    = (const float*)d_in[7];
  const float* gine_w1    = (const float*)d_in[8];
  const float* gine_b1    = (const float*)d_in[9];
  const float* gine_gamma = (const float*)d_in[10];
  const float* gine_beta  = (const float*)d_in[11];
  const float* gine_w2    = (const float*)d_in[12];
  const float* gine_b2    = (const float*)d_in[13];
  const float* gat_wl     = (const float*)d_in[14];
  const float* gat_bl     = (const float*)d_in[15];
  const float* gat_wr     = (const float*)d_in[16];
  const float* gat_br     = (const float*)d_in[17];
  const float* gat_we     = (const float*)d_in[18];
  const float* gat_att    = (const float*)d_in[19];
  const float* gat_bias   = (const float*)d_in[20];
  const float* n0_gamma   = (const float*)d_in[21];
  const float* n0_beta    = (const float*)d_in[22];
  const float* n1_gamma   = (const float*)d_in[23];
  const float* n1_beta    = (const float*)d_in[24];
  const float* skip0      = (const float*)d_in[25];
  const float* skip1      = (const float*)d_in[26];
  const int*   edge_index = (const int*)d_in[31];

  const int N = in_sizes[0] / INDIM;
  const int E = in_sizes[1] / EDIM;
  const int* src = edge_index;
  const int* dst = edge_index + E;

  // ---- workspace carve-up (256B aligned) ----
  char* ws = (char*)d_ws;
  size_t off = 0;
  auto alloc = [&](size_t bytes) -> void* {
    void* p = ws + off;
    off = (off + bytes + 255) & ~(size_t)255;
    return p;
  };
  float*    h0    = (float*)alloc((size_t)N * Hdim * 4);   // zp -> h0 (out0)
  float*    aggr  = (float*)alloc((size_t)N * Hdim * 4);
  float*    t1    = (float*)alloc((size_t)N * TWOH * 4);
  float*    z2    = (float*)alloc((size_t)N * Hdim * 4);   // z2 -> h1pre -> h1 (out1)
  float*    xl    = (float*)alloc((size_t)N * HH * 4);
  float*    xr    = (float*)alloc((size_t)N * HH * 4);
  float*    accum = (float*)alloc((size_t)N * HH * 4);
  float*    h2    = (float*)alloc((size_t)N * Hdim * 4);   // h2pre -> h2 (out2)
  float*    logits= (float*)alloc((size_t)(E + N) * HEADS * 4);
  unsigned* lmaxu = (unsigned*)alloc((size_t)N * HEADS * 4);
  float*    denom = (float*)alloc((size_t)N * HEADS * 4);
  float*    stats = (float*)alloc(2 * TWOH * 4);
  float*    mv    = (float*)alloc(2 * TWOH * 4);
  float*    sum6  = (float*)alloc(64);
  float*    eemean= (float*)alloc(HH * 4);
  _Float16* zh    = (_Float16*)alloc((size_t)N * Hdim * 2);
  _Float16* r1h   = (_Float16*)alloc((size_t)N * TWOH * 2);
  _Float16* h1h   = (_Float16*)alloc((size_t)N * Hdim * 2);
  _Float16* w1h   = (_Float16*)alloc((size_t)Hdim * TWOH * 2);   // packed fragments
  _Float16* w2h   = (_Float16*)alloc((size_t)TWOH * Hdim * 2);
  _Float16* wlh   = (_Float16*)alloc((size_t)Hdim * HH * 2);
  _Float16* wrh   = (_Float16*)alloc((size_t)Hdim * HH * 2);
  (void)ws_size; (void)n_in; (void)out_size;

  const int T = 256;
  const int GS = 2048;   // grid-stride fills

  // ---- init accumulators (must re-zero every call) ----
  fill_f32_kernel<<<GS, T, 0, stream>>>(aggr, 0.f, (size_t)N * Hdim);
  fill_f32_kernel<<<GS, T, 0, stream>>>(accum, 0.f, (size_t)N * HH);
  fill_f32_kernel<<<GS, T, 0, stream>>>(denom, 0.f, (size_t)N * HEADS);
  fill_u32_kernel<<<GS, T, 0, stream>>>(lmaxu, ENC_NEG_INF, (size_t)N * HEADS);
  fill_f32_kernel<<<1, 64, 0, stream>>>(sum6, 0.f, 8);

  // ---- pack weights into f16 WMMA B-fragment layout ----
  pack_b_kernel<<<GS, T, 0, stream>>>(gine_w1, w1h, Hdim, TWOH);
  pack_b_kernel<<<GS, T, 0, stream>>>(gine_w2, w2h, TWOH, Hdim);
  pack_b_kernel<<<GS, T, 0, stream>>>(gat_wl, wlh, Hdim, HH);
  pack_b_kernel<<<GS, T, 0, stream>>>(gat_wr, wrh, Hdim, HH);

  // ---- mean(edge_attr) -> self-loop edge embedding ----
  ea_sum_kernel<<<1024, T, 0, stream>>>(edge_attr, sum6, E);
  ee_mean_kernel<<<nb(HH, T), T, 0, stream>>>(sum6, gat_we, eemean, E);

  // ---- input projection + BN + relu -> h0 (out0) ----
  ip_proj_kernel<<<nb((size_t)N * Hdim, T), T, 0, stream>>>(x, ip_w, ip_b, h0, N);
  fill_f32_kernel<<<1, T, 0, stream>>>(stats, 0.f, 2 * Hdim);
  col_stats_kernel<<<nb((size_t)Hdim * nb(N, 512), T), T, 0, stream>>>(h0, stats, N, Hdim, 512);
  bn_finalize_kernel<<<1, T, 0, stream>>>(stats, mv, N, Hdim);
  bn_apply_kernel<<<nb((size_t)N * Hdim, T), T, 0, stream>>>(h0, h0, nullptr, mv, ip_gamma,
                                                             ip_beta, (size_t)N * Hdim, Hdim, 1);

  // ---- GINEConv: fused edge message + scatter, then WMMA MLP ----
  gine_scatter_kernel<<<E, Hdim, 0, stream>>>(h0, edge_attr, gine_ew, gine_eb, src, dst, aggr, E);
  add_f16_kernel<<<nb((size_t)N * Hdim, T), T, 0, stream>>>(h0, aggr, zh, (size_t)N * Hdim);

  {
    int waves = ((N + 15) / 16) * (TWOH >> 5);
    wmma_gemm_kernel<<<nb((size_t)waves * 32, T), T, 0, stream>>>(zh, w1h, gine_b1, t1, N, Hdim, TWOH);
  }
  fill_f32_kernel<<<1, T, 0, stream>>>(stats, 0.f, 2 * TWOH);
  col_stats_kernel<<<nb((size_t)TWOH * nb(N, 512), T), T, 0, stream>>>(t1, stats, N, TWOH, 512);
  bn_finalize_kernel<<<1, T, 0, stream>>>(stats, mv, N, TWOH);
  bn_apply_kernel<<<nb((size_t)N * TWOH, T), T, 0, stream>>>(t1, nullptr, r1h, mv, gine_gamma,
                                                             gine_beta, (size_t)N * TWOH, TWOH, 1);
  {
    int waves = ((N + 15) / 16) * (Hdim >> 5);
    wmma_gemm_kernel<<<nb((size_t)waves * 32, T), T, 0, stream>>>(r1h, w2h, gine_b2, z2, N, TWOH, Hdim);
  }
  skip0_kernel<<<nb((size_t)N * Hdim, T), T, 0, stream>>>(h0, z2, skip0, (size_t)N * Hdim);
  fill_f32_kernel<<<1, T, 0, stream>>>(stats, 0.f, 2 * Hdim);
  col_stats_kernel<<<nb((size_t)Hdim * nb(N, 512), T), T, 0, stream>>>(z2, stats, N, Hdim, 512);
  bn_finalize_kernel<<<1, T, 0, stream>>>(stats, mv, N, Hdim);
  bn_apply_kernel<<<nb((size_t)N * Hdim, T), T, 0, stream>>>(z2, z2, h1h, mv, n0_gamma, n0_beta,
                                                             (size_t)N * Hdim, Hdim, 0);   // z2 = h1 (out1)

  // ---- GATv2: WMMA projections then 3-pass attention ----
  {
    int waves = ((N + 15) / 16) * (HH >> 5);
    wmma_gemm_kernel<<<nb((size_t)waves * 32, T), T, 0, stream>>>(h1h, wlh, gat_bl, xl, N, Hdim, HH);
    wmma_gemm_kernel<<<nb((size_t)waves * 32, T), T, 0, stream>>>(h1h, wrh, gat_br, xr, N, Hdim, HH);
  }
  gat_logits_kernel<<<E, 128, 0, stream>>>(xl, xr, edge_attr, gat_we, gat_att, eemean,
                                           src, dst, logits, lmaxu, E, 0, 0);
  gat_logits_kernel<<<N, 128, 0, stream>>>(xl, xr, edge_attr, gat_we, gat_att, eemean,
                                           src, dst, logits, lmaxu, N, 1, E);
  gat_softmax_kernel<<<nb((size_t)(E + N) * HEADS, T), T, 0, stream>>>(logits, lmaxu, denom, dst, E, N);
  gat_aggregate_kernel<<<E, Hdim, 0, stream>>>(xl, logits, denom, src, dst, accum, E, 0, 0);
  gat_aggregate_kernel<<<N, Hdim, 0, stream>>>(xl, logits, denom, src, dst, accum, N, 1, E);
  gat_combine_kernel<<<nb((size_t)N * Hdim, T), T, 0, stream>>>(accum, z2, gat_bias, skip1, h2, N);
  fill_f32_kernel<<<1, T, 0, stream>>>(stats, 0.f, 2 * Hdim);
  col_stats_kernel<<<nb((size_t)Hdim * nb(N, 512), T), T, 0, stream>>>(h2, stats, N, Hdim, 512);
  bn_finalize_kernel<<<1, T, 0, stream>>>(stats, mv, N, Hdim);
  bn_apply_kernel<<<nb((size_t)N * Hdim, T), T, 0, stream>>>(h2, h2, nullptr, mv, n1_gamma, n1_beta,
                                                             (size_t)N * Hdim, Hdim, 0);

  // ---- final = (out0 + out1 + out2) / 3 ----
  final_avg_kernel<<<nb((size_t)N * Hdim, T), T, 0, stream>>>(h0, z2, h2, (float*)d_out,
                                                              (size_t)N * Hdim);
}
